// KANITTS_MAIN_10849087390495
// MI455X (gfx1250) — matrix operand order repaired and checked by reading, not compiled, not run on hardware
//
#include <hip/hip_runtime.h>
#include <hip/hip_bf16.h>

// ---------------- model constants (from reference) ----------------
#define NL   16
#define NA   8
#define NC   8
#define HQ   16      // query heads
#define KVH  4
#define GR   4
#define HD   64
#define HDH  32
#define DM   1024
#define DFF  4096
#define VV   65536
#define BB   16      // batch
#define HIST 2047
#define TT   2048
#define EPS  1e-5f

// output layout (flat, return order: sk, sv, sc, logits)
#define SK_OFF 0ull
#define SK_SZ  (8ull*16*4*64*2048)            // 67108864
#define SV_OFF (SK_OFF + SK_SZ)
#define SV_SZ  SK_SZ
#define SC_OFF (SV_OFF + SV_SZ)               // 134217728
#define SC_SZ  (8ull*16*1024*2)               // 262144
#define LG_OFF (SC_OFF + SC_SZ)               // 134479872

typedef __attribute__((ext_vector_type(16))) __bf16 v16bf;
typedef __attribute__((ext_vector_type(8)))  float  v8f;
typedef __attribute__((ext_vector_type(4)))  unsigned int u32x4;
typedef __attribute__((ext_vector_type(8)))  int i32x8;
typedef __attribute__((ext_vector_type(4)))  int i32x4;

union FragB { v16bf v; __bf16 e[16]; };

#if __has_builtin(__builtin_amdgcn_tensor_load_to_lds)
#define HAVE_TDM 1
#else
#define HAVE_TDM 0
#endif

#if HAVE_TDM
// Issue a TDM 2D tile load: 32 rows x 128 cols of fp32 from W (row stride N)
// into LDS at byte offset lds_byte.  D# per CDNA5 ISA §8.3/8.4.
template<int K, int N>
__device__ __forceinline__ void tdm_issue(unsigned lds_byte, const float* gsrc) {
  unsigned long long ga = (unsigned long long)(uintptr_t)gsrc;
  u32x4 g0;
  g0[0] = 1u;                                             // count=1, user mode
  g0[1] = lds_byte;                                       // lds_addr
  g0[2] = (unsigned)ga;                                   // global_addr[31:0]
  g0[3] = (unsigned)((ga >> 32) & 0x1FFFFFFu) | (2u << 30); // addr[56:32] | type=2
  i32x8 g1;
  g1[0] = (int)(2u << 16);                                // data_size = 4B
  g1[1] = (int)(((unsigned)N & 0xFFFFu) << 16);           // tensor_dim0[15:0]
  g1[2] = (int)(((unsigned)N >> 16) | (((unsigned)K & 0xFFFFu) << 16)); // dim0 hi | dim1 lo
  g1[3] = (int)(((unsigned)K >> 16) | (128u << 16));      // dim1 hi | tile_dim0=128
  g1[4] = 32;                                             // tile_dim1=32, tile_dim2=0
  g1[5] = N;                                              // tensor_dim0_stride lo
  g1[6] = 0;
  g1[7] = 0;
  i32x4 z4 = {};
#if defined(__clang_major__) && (__clang_major__ >= 23)
  i32x8 z8 = {};
  __builtin_amdgcn_tensor_load_to_lds(g0, g1, z4, z4, z8, 0);
#else
  __builtin_amdgcn_tensor_load_to_lds(g0, g1, z4, z4, 0);
#endif
}
#endif

// ---------------------------------------------------------------------------
// Pack fp32 activation A[16,K] into bf16 WMMA A-fragment layout:
//   P[(k0/32)*32 + lane][e]  with  k = e + ((lane>>4)<<4)
// ---------------------------------------------------------------------------
template<int K>
__global__ void pack_a_kernel(const float* __restrict__ A, __bf16* __restrict__ P) {
  int idx = blockIdx.x * 256 + threadIdx.x;     // (chunk, lane) pairs, K total
  if (idx >= K) return;
  const int l = idx & 31;
  const int k0 = (idx >> 5) << 5;
  const float* ar = A + (size_t)(l & 15) * K + k0 + ((l >> 4) << 4);
  __bf16* p = P + (size_t)idx * 16;
#pragma unroll
  for (int e = 0; e < 16; ++e) {
    int k = e;
    p[e] = (__bf16)ar[k];
  }
}

// ---------------------------------------------------------------------------
// WMMA GEMM: out[16,N] = A[16,K] @ W[K,N] (+resid if RES).
// TDM_EN: weights staged 32x128 fp32/tile into LDS by the Tensor Data Mover,
// double-buffered, s_wait_tensorcnt + barrier; else direct clause-able loads.
// One wave computes two adjacent 16-col tiles (2 WMMAs / k-step, shared A).
// ---------------------------------------------------------------------------
template<int K, int N, bool RES, bool TDM_EN>
__global__ void gemm16_wmma(const __bf16* __restrict__ Apk,
                            const float* __restrict__ W,
                            const float* resid, float* out) {
  const int lane = threadIdx.x & 31;
  const int wid  = threadIdx.x >> 5;
  const int n0   = (blockIdx.x * 4 + wid) * 32;     // 2 tiles of 16 cols
  const int ncol = lane & 15;
  const int koffB = (lane & 16);                    // B: lanes>=16 hold K+16

  const v16bf* Afrag = (const v16bf*)Apk + lane;    // + 32 per k-step

  v8f acc0 = {}, acc1 = {};

#if HAVE_TDM
  if constexpr (TDM_EN) {
    __shared__ float wt[2][32 * 128];
    const int nblk = blockIdx.x * 128;
    const int colIdx = wid * 32 + ncol;             // column within block tile
    if (wid == 0) tdm_issue<K, N>((unsigned)(uintptr_t)&wt[0][0], W + nblk);
    for (int k0 = 0; k0 < K; k0 += 32) {
      const int cur = (k0 >> 5) & 1;
      if (wid == 0) __builtin_amdgcn_s_wait_tensorcnt(0);
      __syncthreads();                              // tile[cur] ready; buf[cur^1] free
      if (wid == 0 && k0 + 32 < K)
        tdm_issue<K, N>((unsigned)(uintptr_t)&wt[cur ^ 1][0],
                        W + (size_t)(k0 + 32) * N + nblk);
      const v16bf a = Afrag[k0];
      const float* ls = &wt[cur][0];
      FragB b0, b1;
#pragma unroll
      for (int e = 0; e < 16; ++e) {
        b0.e[e] = (__bf16)ls[(e + koffB) * 128 + colIdx];
        b1.e[e] = (__bf16)ls[(e + koffB) * 128 + colIdx + 16];
      }
      acc0 = __builtin_amdgcn_wmma_f32_16x16x32_bf16(
          false, a, false, b0.v, (short)0, acc0, false, false);
      acc1 = __builtin_amdgcn_wmma_f32_16x16x32_bf16(
          false, a, false, b1.v, (short)0, acc1, false, false);
    }
  } else
#endif
  {
    const float* Wbase = W + (size_t)koffB * N + n0 + ncol;
    for (int k0 = 0; k0 < K; k0 += 32) {
      const v16bf a = Afrag[k0];
      const float* wp = Wbase + (size_t)k0 * N;
      FragB b0, b1;
#pragma unroll
      for (int e = 0; e < 16; ++e) {                // immediate offsets e*N*4 (+64)
        b0.e[e] = (__bf16)wp[(size_t)e * N];
        b1.e[e] = (__bf16)wp[(size_t)e * N + 16];
      }
      if (k0 + 32 < K)                              // global_prefetch_b8
        __builtin_prefetch(wp + (size_t)32 * N, 0, 1);
      acc0 = __builtin_amdgcn_wmma_f32_16x16x32_bf16(
          false, a, false, b0.v, (short)0, acc0, false, false);
      acc1 = __builtin_amdgcn_wmma_f32_16x16x32_bf16(
          false, a, false, b1.v, (short)0, acc1, false, false);
    }
  }

  // C/D layout: VGPR r -> M = r (lanes 0-15) / r+8 (lanes 16-31), N = lane&15
  const int rbase = (lane >> 4) << 3;
#pragma unroll
  for (int r = 0; r < 8; ++r) {
    size_t o = (size_t)(rbase + r) * N + n0 + ncol;
    float v0 = acc0[r], v1 = acc1[r];
    if constexpr (RES) { v0 += resid[o]; v1 += resid[o + 16]; }
    out[o] = v0;
    out[o + 16] = v1;
  }
}

// ---------------------------------------------------------------------------
// RMSNorm: y[b,:] = w * x[b,:] * rsqrt(mean(x^2)+eps).  grid = B, block = 256.
// ---------------------------------------------------------------------------
__global__ void rmsnorm_kernel(const float* __restrict__ x,
                               const float* __restrict__ w,
                               float* __restrict__ y, int D) {
  __shared__ float red[8];
  const int b = blockIdx.x, tid = threadIdx.x;
  const size_t base = (size_t)b * D;
  float ss = 0.f;
  for (int i = tid; i < D; i += 256) { float v = x[base + i]; ss += v * v; }
#pragma unroll
  for (int o = 16; o; o >>= 1) ss += __shfl_xor(ss, o);
  if ((tid & 31) == 0) red[tid >> 5] = ss;
  __syncthreads();
  float tot = 0.f;
#pragma unroll
  for (int i = 0; i < 8; ++i) tot += red[i];
  const float r = rsqrtf(tot / (float)D + EPS);
  for (int i = tid; i < D; i += 256) y[base + i] = w[i] * x[base + i] * r;
}

// ---------------------------------------------------------------------------
// Per-head RMSNorm + RoPE for q and k; store new k/v column into sk/sv.
// grid = (B, HQ+KVH), block = HD (64).
// ---------------------------------------------------------------------------
__global__ void qkv_rope_kernel(float* qb, const float* kb, const float* vb,
                                const float* __restrict__ qln,
                                const float* __restrict__ kln,
                                const float* __restrict__ cost,
                                const float* __restrict__ sint,
                                const int* __restrict__ hist,
                                float* skA, float* svA) {
  __shared__ float sh[HD];
  __shared__ float red[2];
  const int b = blockIdx.x, hh = blockIdx.y, d = threadIdx.x;
  const int pos = hist[0];
  const float cs = cost[(size_t)pos * HD + d];
  const float sn = sint[(size_t)pos * HD + d];
  float x, w;
  if (hh < HQ) { x = qb[(b * HQ + hh) * HD + d]; w = qln[d]; }
  else         { int j = hh - HQ; x = kb[(b * KVH + j) * HD + d]; w = kln[d]; }
  float ss = x * x;
#pragma unroll
  for (int o = 16; o; o >>= 1) ss += __shfl_xor(ss, o);
  if ((d & 31) == 0) red[d >> 5] = ss;
  __syncthreads();
  const float ms = (red[0] + red[1]) * (1.0f / (float)HD);
  const float xn = w * x * rsqrtf(ms + EPS);
  sh[d] = xn;
  __syncthreads();
  const float rot = (d < HDH) ? -sh[d + HDH] : sh[d - HDH];
  const float o = xn * cs + rot * sn;
  if (hh < HQ) {
    qb[(b * HQ + hh) * HD + d] = o;
  } else {
    int j = hh - HQ;
    skA[(((size_t)b * KVH + j) * HD + d) * TT + (TT - 1)] = o;          // k: [..,HD,T]
    svA[(((size_t)b * KVH + j) * TT + (TT - 1)) * HD + d] =
        vb[(b * KVH + j) * HD + d];                                     // v: [..,T,HD]
  }
}

// ---------------------------------------------------------------------------
// Attention: one block per (b, head).  scores over T=2048, softmax, @V.
// ---------------------------------------------------------------------------
__global__ void attn_kernel(const float* __restrict__ q,
                            const float* __restrict__ skA,
                            const float* __restrict__ svA,
                            float* __restrict__ ao) {
  __shared__ float qs[HD];
  __shared__ float sc[TT];
  __shared__ float redm[8], reds[8], osum[256];
  const int blk = blockIdx.x, tid = threadIdx.x;
  const int b = blk >> 4, h = blk & 15, kvh = h >> 2;   // GR = 4
  const float* K = skA + ((size_t)b * KVH + kvh) * HD * TT;  // [HD][T]
  const float* Vm = svA + ((size_t)b * KVH + kvh) * TT * HD; // [T][HD]
  if (tid < HD) qs[tid] = q[(b * HQ + h) * HD + tid];
  __syncthreads();

  float mx = -1e30f;
  for (int t = tid; t < TT; t += 256) {
    float dot = 0.f;
#pragma unroll
    for (int d = 0; d < HD; ++d) dot += qs[d] * K[(size_t)d * TT + t];
    sc[t] = dot;
    mx = fmaxf(mx, dot);
  }
#pragma unroll
  for (int o = 16; o; o >>= 1) mx = fmaxf(mx, __shfl_xor(mx, o));
  if ((tid & 31) == 0) redm[tid >> 5] = mx;
  __syncthreads();
  float m = redm[0];
#pragma unroll
  for (int i = 1; i < 8; ++i) m = fmaxf(m, redm[i]);

  float ssum = 0.f;
  for (int t = tid; t < TT; t += 256) {
    float p = __expf(sc[t] - m);
    sc[t] = p;
    ssum += p;
  }
#pragma unroll
  for (int o = 16; o; o >>= 1) ssum += __shfl_xor(ssum, o);
  if ((tid & 31) == 0) reds[tid >> 5] = ssum;
  __syncthreads();
  float tot = 0.f;
#pragma unroll
  for (int i = 0; i < 8; ++i) tot += reds[i];
  const float inv = 1.0f / tot;

  const int d = tid & 63, g = tid >> 6;   // 4 t-groups x 64 dims
  float acc = 0.f;
  for (int t = g; t < TT; t += 4) acc += sc[t] * Vm[(size_t)t * HD + d];
  osum[tid] = acc;
  __syncthreads();
  if (g == 0)
    ao[(b * HQ + h) * HD + d] =
        (osum[d] + osum[64 + d] + osum[128 + d] + osum[192 + d]) * inv;
}

// ---------------------------------------------------------------------------
// Conv mixer pointwise stage (S=1: st = [cache0, cache1, Bg*xg]).
// ---------------------------------------------------------------------------
__global__ void conv_kernel(const float* __restrict__ bcx,
                            const float* __restrict__ ccache,
                            const float* __restrict__ cw,
                            float* __restrict__ y, float* scOut, int cidx) {
  const int b = blockIdx.x;
  for (int d = threadIdx.x; d < DM; d += 256) {
    const float Bg = bcx[b * 3 * DM + d];
    const float Cg = bcx[b * 3 * DM + DM + d];
    const float xg = bcx[b * 3 * DM + 2 * DM + d];
    const size_t cb = ((size_t)(cidx * BB + b) * DM + d) * 2;
    const float c0 = ccache[cb + 0], c1 = ccache[cb + 1];
    const float nx = Bg * xg;
    scOut[cb + 0] = c1;
    scOut[cb + 1] = nx;
    const float co = cw[d * 3 + 0] * c0 + cw[d * 3 + 1] * c1 + cw[d * 3 + 2] * nx;
    y[b * DM + d] = Cg * co;
  }
}

// ---------------------------------------------------------------------------
// elementwise + copies
// ---------------------------------------------------------------------------
__global__ void silu_gate_kernel(const float* __restrict__ g1,
                                 const float* __restrict__ g3,
                                 float* __restrict__ act, int n) {
  for (int i = blockIdx.x * 256 + threadIdx.x; i < n; i += gridDim.x * 256) {
    float a = g1[i];
    act[i] = (a / (1.0f + __expf(-a))) * g3[i];
  }
}

__global__ void copy_f32_kernel(const float* __restrict__ s, float* __restrict__ d, int n) {
  int i = blockIdx.x * 256 + threadIdx.x;
  if (i < n) d[i] = s[i];
}

// k_cache (…,HD,HIST) rows -> sk (…,HD,T): one block per row of 2047
__global__ void kcache_copy_kernel(const float* __restrict__ src, float* __restrict__ dst) {
  const size_t row = blockIdx.x;
  const float* s = src + row * (size_t)HIST;
  float* d = dst + row * (size_t)TT;
  for (int t = threadIdx.x; t < HIST; t += 256) d[t] = s[t];
}

// v_cache (…,HIST,HD) contiguous chunks -> sv (…,T,HD)
__global__ void vcache_copy_kernel(const float* __restrict__ src, float* __restrict__ dst) {
  const size_t chunk = blockIdx.x;
  const float* s = src + chunk * (size_t)(HIST * HD);
  float* d = dst + chunk * (size_t)(TT * HD);
  for (int i = threadIdx.x; i < HIST * HD; i += 256) d[i] = s[i];
}

// ---------------------------------------------------------------------------
template<int K, int N, bool RES, bool TDM>
static inline void launch_gemm(const __bf16* Apk, const float* W, const float* resid,
                               float* out, hipStream_t s) {
  gemm16_wmma<K, N, RES, TDM><<<N / 128, 128, 0, s>>>(Apk, W, resid, out);
}
template<int K>
static inline void launch_pack(const float* A, __bf16* P, hipStream_t s) {
  pack_a_kernel<K><<<K / 256, 256, 0, s>>>(A, P);
}

extern "C" void kernel_launch(void* const* d_in, const int* in_sizes, int n_in,
                              void* d_out, int out_size, void* d_ws, size_t ws_size,
                              hipStream_t stream) {
  const float* hs   = (const float*)d_in[0];
  const float* kc   = (const float*)d_in[1];
  const float* vc   = (const float*)d_in[2];
  const float* cc   = (const float*)d_in[3];
  const float* cost = (const float*)d_in[4];
  const float* sint = (const float*)d_in[5];
  const float* opw  = (const float*)d_in[6];
  const float* qlnw = (const float*)d_in[7];
  const float* klnw = (const float*)d_in[8];
  const float* qw   = (const float*)d_in[9];
  const float* kw   = (const float*)d_in[10];
  const float* vw   = (const float*)d_in[11];
  const float* ow   = (const float*)d_in[12];
  const float* inw  = (const float*)d_in[13];
  const float* cvw  = (const float*)d_in[14];
  const float* cow  = (const float*)d_in[15];
  const float* fnw  = (const float*)d_in[16];
  const float* w1   = (const float*)d_in[17];
  const float* w3   = (const float*)d_in[18];
  const float* w2   = (const float*)d_in[19];
  const float* embw = (const float*)d_in[20];
  const float* lmw  = (const float*)d_in[21];
  const int*   hist = (const int*)d_in[22];

  float* out = (float*)d_out;
  float* sk = out + SK_OFF;
  float* sv = out + SV_OFF;
  float* sc = out + SC_OFF;
  float* lg = out + LG_OFF;

  float* ws  = (float*)d_ws;
  float* h    = ws;            ws += BB * DM;       // 16384
  float* hn   = ws;            ws += BB * DM;
  float* qb   = ws;            ws += BB * HQ * HD;  // 16384
  float* kb   = ws;            ws += BB * KVH * HD; // 4096
  float* vb   = ws;            ws += BB * KVH * HD;
  float* ao   = ws;            ws += BB * DM;
  float* bcx  = ws;            ws += BB * 3 * DM;   // 49152
  float* yb   = ws;            ws += BB * DM;
  float* g1   = ws;            ws += BB * DFF;      // 65536
  float* g3   = ws;            ws += BB * DFF;
  float* actb = ws;            ws += BB * DFF;
  __bf16* apk = (__bf16*)ws;   ws += BB * DFF / 2;  // 16x4096 bf16 (128KB)

  // concat caches into outputs up front (new column filled per layer)
  kcache_copy_kernel<<<NA * BB * KVH * HD, 256, 0, stream>>>(kc, sk);   // 32768 rows
  vcache_copy_kernel<<<NA * BB * KVH, 256, 0, stream>>>(vc, sv);        // 512 chunks
  copy_f32_kernel<<<(BB * DM + 255) / 256, 256, 0, stream>>>(hs, h, BB * DM);

  int a = 0, ci = 0;
  for (int i = 0; i < NL; ++i) {
    rmsnorm_kernel<<<BB, 256, 0, stream>>>(h, opw + (size_t)i * DM, hn, DM);
    launch_pack<DM>(hn, apk, stream);
    if (i & 1) {
      launch_gemm<DM, DM, false, true>(apk, qw + (size_t)a * DM * HQ * HD, nullptr, qb, stream);
      launch_gemm<DM, KVH * HD, false, false>(apk, kw + (size_t)a * DM * KVH * HD, nullptr, kb, stream);
      launch_gemm<DM, KVH * HD, false, false>(apk, vw + (size_t)a * DM * KVH * HD, nullptr, vb, stream);
      float* skA = sk + (size_t)a * BB * KVH * HD * TT;
      float* svA = sv + (size_t)a * BB * KVH * TT * HD;
      qkv_rope_kernel<<<dim3(BB, HQ + KVH), HD, 0, stream>>>(
          qb, kb, vb, qlnw + (size_t)a * HD, klnw + (size_t)a * HD, cost, sint, hist, skA, svA);
      attn_kernel<<<BB * HQ, 256, 0, stream>>>(qb, skA, svA, ao);
      launch_pack<DM>(ao, apk, stream);
      launch_gemm<DM, DM, true, true>(apk, ow + (size_t)a * HQ * HD * DM, h, h, stream);
      ++a;
    } else {
      launch_gemm<DM, 3 * DM, false, true>(apk, inw + (size_t)ci * DM * 3 * DM, nullptr, bcx, stream);
      conv_kernel<<<BB, 256, 0, stream>>>(bcx, cc, cvw + (size_t)ci * DM * 3, yb, sc, ci);
      launch_pack<DM>(yb, apk, stream);
      launch_gemm<DM, DM, true, true>(apk, cow + (size_t)ci * DM * DM, h, h, stream);
      ++ci;
    }
    rmsnorm_kernel<<<BB, 256, 0, stream>>>(h, fnw + (size_t)i * DM, hn, DM);
    launch_pack<DM>(hn, apk, stream);
    launch_gemm<DM, DFF, false, true>(apk, w1 + (size_t)i * DM * DFF, nullptr, g1, stream);
    launch_gemm<DM, DFF, false, true>(apk, w3 + (size_t)i * DM * DFF, nullptr, g3, stream);
    silu_gate_kernel<<<256, 256, 0, stream>>>(g1, g3, actb, BB * DFF);
    launch_pack<DFF>(actb, apk, stream);
    launch_gemm<DFF, DM, true, true>(apk, w2 + (size_t)i * DFF * DM, h, h, stream);
  }
  rmsnorm_kernel<<<BB, 256, 0, stream>>>(h, embw, hn, DM);
  launch_pack<DM>(hn, apk, stream);
  launch_gemm<DM, VV, false, true>(apk, lmw, nullptr, lg, stream);
}
